// NCP_65532611002703
// MI455X (gfx1250) — matrix operand order repaired and checked by reading, not compiled, not run on hardware
//
#include <hip/hip_runtime.h>
#include <math.h>

typedef __attribute__((ext_vector_type(16))) _Float16 v16h;
typedef __attribute__((ext_vector_type(8)))  float    v8f;
typedef __attribute__((ext_vector_type(4)))  float    vf4;   // clang vector (not HIP_vector_type)

// Problem constants
#define BATCH   256
#define TSTEPS  512
#define NFEAT   256
#define H1      37
#define H2      24
#define H3      3
#define NPROJ4  148          // 4 * H1
#define NPAD    160          // padded to 10 tiles of 16
#define NTILES  10
#define KSTEPS  8            // 256 / 32
// Packed B: NTILES * KSTEPS * 32 lanes * 16 halves
#define BPACK_HALVES (NTILES * KSTEPS * 32 * 16)   // 40960
#define BPACK_BYTES  (BPACK_HALVES * 2)            // 81920 (128B aligned)

__device__ __forceinline__ float sigmoidf_fast(float v) {
    return 1.0f / (1.0f + __expf(-v));
}

// ---------------------------------------------------------------------------
// Kernel 1: pack layer-0 input-part weights (masked ff1/ff2, dense ta/tb)
// into f16 WMMA B-matrix tiles. B layout (16-bit, 32x16 KxN, wave32):
//   half i -> K = kstep*32 + (i>>3)*16 + (lane>>4)*8 + (i&7)   (N = lane & 15)
// Flat index: (((n*KSTEPS + k)*32) + lane)*16 + i
// ---------------------------------------------------------------------------
__global__ void ncp_pack_b(const float* __restrict__ ff1,
                           const float* __restrict__ ff2,
                           const float* __restrict__ ta,
                           const float* __restrict__ tb,
                           const float* __restrict__ mask0,
                           _Float16* __restrict__ Bp) {
    int idx = blockIdx.x * blockDim.x + threadIdx.x;
    if (idx >= BPACK_HALVES) return;
    int i    = idx & 15;
    int lane = (idx >> 4) & 31;
    int k    = (idx >> 9) & 7;
    int n    = idx >> 12;
    int col  = n * 16 + (lane & 15);
    int K    = k * 32 + (i >> 3) * 16 + (lane >> 4) * 8 + (i & 7);
    float w = 0.0f;
    if (col < NPROJ4) {
        int proj = col / H1;
        int j    = col - proj * H1;
        const float* W = (proj == 0) ? ff1 : (proj == 1) ? ff2 : (proj == 2) ? ta : tb;
        w = W[j * (NFEAT + H1) + K];               // layer-0 weight row stride = 293
        if (proj < 2) w *= mask0[j * (NFEAT + H1) + K];
    }
    Bp[idx] = (_Float16)w;
}

// ---------------------------------------------------------------------------
// Kernel 2: hoisted input GEMM  Xp[t][b][0..147] = x[b,t,:] @ Wsel.T
// One wave per 16-row M-tile (16 consecutive b at fixed t). A fragments for
// all 8 k-steps built once, reused across all 10 N-tiles -> 80 v_wmma per
// wave, f32 accumulation.
//
// Cache policy (192 MB L2): x (134 MB) is read exactly once -> non-temporal
// loads so it does not evict Xp. Xp (84 MB) fits in L2 and is re-read by the
// scan kernel -> default RT stores/loads keep it L2-resident. Bp (80 KB) is
// reused by all 8192 waves -> default RT, stays cache-hot.
//
// A layout (16-bit, 16x32 MxK): lane m = lane&15, khalf = lane>>4,
//   half i -> K = (i>>3)*16 + khalf*8 + (i&7).
// ---------------------------------------------------------------------------
__global__ __launch_bounds__(128) void ncp_in_gemm(const float* __restrict__ x,
                                                   const _Float16* __restrict__ Bp,
                                                   float* __restrict__ Xp) {
    const int lane  = threadIdx.x & 31;
    const int wave  = threadIdx.x >> 5;
    const int g     = blockIdx.x * 4 + wave;     // tile id: 0 .. 512*16-1
    const int t     = g >> 4;
    const int b0    = (g & 15) << 4;
    const int m     = lane & 15;
    const int khalf = lane >> 4;

    const float* xr = x + (((size_t)(b0 + m)) * TSTEPS + t) * NFEAT;

    v16h a[KSTEPS];
#pragma unroll
    for (int k = 0; k < KSTEPS; ++k) {
        const float* p0 = xr + k * 32 + khalf * 8;        // K = k*32 + khalf*8 + 0..7
        const float* p1 = p0 + 16;                        // K = ... + 16
        vf4 u0 = __builtin_nontemporal_load((const vf4*)(p0));
        vf4 u1 = __builtin_nontemporal_load((const vf4*)(p0 + 4));
        vf4 w0 = __builtin_nontemporal_load((const vf4*)(p1));
        vf4 w1 = __builtin_nontemporal_load((const vf4*)(p1 + 4));
        v16h av;
#pragma unroll
        for (int i = 0; i < 4; ++i) {
            av[i]      = (_Float16)u0[i];
            av[4 + i]  = (_Float16)u1[i];
            av[8 + i]  = (_Float16)w0[i];
            av[12 + i] = (_Float16)w1[i];
        }
        a[k] = av;
    }

    for (int n = 0; n < NTILES; ++n) {
        v8f c = {};
#pragma unroll
        for (int k = 0; k < KSTEPS; ++k) {
            v16h b = *(const v16h*)(Bp + (((size_t)(n * KSTEPS + k) * 32) + lane) * 16);
            c = __builtin_amdgcn_wmma_f32_16x16x32_f16(
                    false, a[k], false, b, (short)0, c, false, false);
        }
        // D layout: c[r] = D[M = r + 8*khalf][N = lane&15]
        const int col = n * 16 + m;
        const size_t base = ((size_t)t * BATCH + b0) * NPAD + col;
#pragma unroll
        for (int r2 = 0; r2 < 8; ++r2) {
            Xp[base + (size_t)(r2 + 8 * khalf) * NPAD] = c[r2];
        }
    }
}

// ---------------------------------------------------------------------------
// Kernel 3: recurrent scan, f32 VALU. 64 blocks x 4 batch rows, thread =
// (row r, slot c); weights (mask-merged) + biases live in LDS for all 512
// steps. Per step: 3 cascaded CfC cells with sync/write/sync on h buffers.
// Xp reads hit L2 (kept resident by NT policy on x above).
// ---------------------------------------------------------------------------
struct ScanArgs {
    const float* w0[8];    // ff1_w, ff1_b, ff2_w, ff2_b, ta_w, ta_b, tb_w, tb_b (layer 0)
    const float* w1[8];    // layer 1
    const float* w2[8];    // layer 2
    const float* mask1;    // [24, 61]
    const float* mask2;    // [3, 27]
    const float* hx;       // [256, 64]
    const float* xp;       // [T][B][NPAD]
    float* out_controls;   // [256, 512, 3]
    float* out_hx;         // [256, 64]
};

#define SROWS 4
#define STHREADS 160        // SROWS * 40

__global__ __launch_bounds__(STHREADS) void ncp_scan(ScanArgs A) {
    __shared__ float W1r[4 * H1 * H1];      // [p][j][k] recurrent part (mask==1 there)
    __shared__ float B1[4 * H1];
    __shared__ float W2[4 * H2 * 61];       // [p][j][k], k over [h1(37) | h2(24)]
    __shared__ float B2[4 * H2];
    __shared__ float W3[4 * H3 * 27];       // [p][j][k], k over [h2(24) | h3(3)]
    __shared__ float B3[4 * H3];
    __shared__ float h1s[SROWS * H1];
    __shared__ float h2s[SROWS * H2];
    __shared__ float h3s[SROWS * H3];

    const int tid = threadIdx.x;
    const int b0  = blockIdx.x * SROWS;

    // Merge weights (masked where the reference masks) into LDS, once.
    for (int i = tid; i < 4 * H1 * H1; i += STHREADS) {
        int p = i / (H1 * H1); int rem = i - p * (H1 * H1);
        int j = rem / H1;      int k = rem - j * H1;
        W1r[i] = A.w0[p * 2][j * (NFEAT + H1) + NFEAT + k];   // recurrent cols 256..292
    }
    for (int i = tid; i < 4 * H2 * 61; i += STHREADS) {
        int p = i / (H2 * 61); int rem = i - p * (H2 * 61);
        int j = rem / 61;      int k = rem - j * 61;
        float w = A.w1[p * 2][j * 61 + k];
        if (p < 2) w *= A.mask1[j * 61 + k];
        W2[i] = w;
    }
    for (int i = tid; i < 4 * H3 * 27; i += STHREADS) {
        int p = i / (H3 * 27); int rem = i - p * (H3 * 27);
        int j = rem / 27;      int k = rem - j * 27;
        float w = A.w2[p * 2][j * 27 + k];
        if (p < 2) w *= A.mask2[j * 27 + k];
        W3[i] = w;
    }
    if (tid < 4 * H1) {
        B1[tid] = A.w0[(tid / H1) * 2 + 1][tid % H1];
    }
    for (int i = tid; i < 4 * H2; i += STHREADS) {
        B2[i] = A.w1[(i / H2) * 2 + 1][i % H2];
    }
    if (tid < 4 * H3) {
        B3[tid] = A.w2[(tid / H3) * 2 + 1][tid % H3];
    }
    // Init hidden state from hx (splits: 37 | 24 | 3).
    for (int i = tid; i < SROWS * 64; i += STHREADS) {
        int bl = i >> 6, f = i & 63;
        float v = A.hx[(size_t)(b0 + bl) * 64 + f];
        if (f < H1)            h1s[bl * H1 + f] = v;
        else if (f < H1 + H2)  h2s[bl * H2 + (f - H1)] = v;
        else                   h3s[bl * H3 + (f - H1 - H2)] = v;
    }
    __syncthreads();

    const int r = tid / 40;
    const int c = tid - r * 40;

    for (int t = 0; t < TSTEPS; ++t) {
        // ---- layer 0 (inter): input part precomputed in Xp ----
        float hn1 = 0.0f;
        if (c < H1) {
            const float* xpr = A.xp + ((size_t)t * BATCH + b0 + r) * NPAD;
            float a0 = B1[c]          + xpr[c];
            float a1 = B1[H1 + c]     + xpr[H1 + c];
            float a2 = B1[2 * H1 + c] + xpr[2 * H1 + c];
            float a3 = B1[3 * H1 + c] + xpr[3 * H1 + c];
            const float* wp = &W1r[c * H1];
            for (int k = 0; k < H1; ++k) {
                float hk = h1s[r * H1 + k];
                a0 = fmaf(hk, wp[k],                a0);
                a1 = fmaf(hk, wp[H1 * H1 + k],      a1);
                a2 = fmaf(hk, wp[2 * H1 * H1 + k],  a2);
                a3 = fmaf(hk, wp[3 * H1 * H1 + k],  a3);
            }
            float ti = sigmoidf_fast(a2 + a3);      // ts = 1.0
            float f1 = tanhf(a0), f2 = tanhf(a1);
            hn1 = f1 + ti * (f2 - f1);
        }
        __syncthreads();
        if (c < H1) h1s[r * H1 + c] = hn1;
        __syncthreads();

        // ---- layer 1 (command): z = [h1_new(37) | h2(24)] ----
        float hn2 = 0.0f;
        if (c < H2) {
            float a0 = B2[c], a1 = B2[H2 + c], a2 = B2[2 * H2 + c], a3 = B2[3 * H2 + c];
            const float* wp = &W2[c * 61];
            for (int k = 0; k < H1; ++k) {
                float zk = h1s[r * H1 + k];
                a0 = fmaf(zk, wp[k],               a0);
                a1 = fmaf(zk, wp[H2 * 61 + k],     a1);
                a2 = fmaf(zk, wp[2 * H2 * 61 + k], a2);
                a3 = fmaf(zk, wp[3 * H2 * 61 + k], a3);
            }
            for (int k = 0; k < H2; ++k) {
                float zk = h2s[r * H2 + k];
                a0 = fmaf(zk, wp[H1 + k],               a0);
                a1 = fmaf(zk, wp[H2 * 61 + H1 + k],     a1);
                a2 = fmaf(zk, wp[2 * H2 * 61 + H1 + k], a2);
                a3 = fmaf(zk, wp[3 * H2 * 61 + H1 + k], a3);
            }
            float ti = sigmoidf_fast(a2 + a3);
            float f1 = tanhf(a0), f2 = tanhf(a1);
            hn2 = f1 + ti * (f2 - f1);
        }
        __syncthreads();
        if (c < H2) h2s[r * H2 + c] = hn2;
        __syncthreads();

        // ---- layer 2 (motor): z = [h2_new(24) | h3(3)] ----
        float hn3 = 0.0f;
        if (c < H3) {
            float a0 = B3[c], a1 = B3[H3 + c], a2 = B3[2 * H3 + c], a3 = B3[3 * H3 + c];
            const float* wp = &W3[c * 27];
            for (int k = 0; k < H2; ++k) {
                float zk = h2s[r * H2 + k];
                a0 = fmaf(zk, wp[k],               a0);
                a1 = fmaf(zk, wp[H3 * 27 + k],     a1);
                a2 = fmaf(zk, wp[2 * H3 * 27 + k], a2);
                a3 = fmaf(zk, wp[3 * H3 * 27 + k], a3);
            }
            for (int k = 0; k < H3; ++k) {
                float zk = h3s[r * H3 + k];
                a0 = fmaf(zk, wp[H2 + k],               a0);
                a1 = fmaf(zk, wp[H3 * 27 + H2 + k],     a1);
                a2 = fmaf(zk, wp[2 * H3 * 27 + H2 + k], a2);
                a3 = fmaf(zk, wp[3 * H3 * 27 + H2 + k], a3);
            }
            float ti = sigmoidf_fast(a2 + a3);
            float f1 = tanhf(a0), f2 = tanhf(a1);
            hn3 = f1 + ti * (f2 - f1);
            // controls: steer=tanh(y0), throttle=sigmoid(y1), brake=sigmoid(y2)
            float outv = (c == 0) ? tanhf(hn3) : sigmoidf_fast(hn3);
            A.out_controls[((size_t)(b0 + r) * TSTEPS + t) * 3 + c] = outv;
        }
        __syncthreads();
        if (c < H3) h3s[r * H3 + c] = hn3;
        __syncthreads();
    }

    // Final hidden state: concat [h1 | h2 | h3] per batch row.
    for (int i = tid; i < SROWS * 64; i += STHREADS) {
        int bl = i >> 6, f = i & 63;
        float v = (f < H1)      ? h1s[bl * H1 + f]
                : (f < H1 + H2) ? h2s[bl * H2 + (f - H1)]
                                : h3s[bl * H3 + (f - H1 - H2)];
        A.out_hx[(size_t)(b0 + bl) * 64 + f] = v;
    }
}

// ---------------------------------------------------------------------------
// d_in order (setup_inputs dict flattened):
//  [0] x [256,512,256]  [1] hx [256,64]
//  [2..9]   layer0: ff1_w, ff1_b, ff2_w, ff2_b, ta_w, ta_b, tb_w, tb_b
//  [10..17] layer1 (same order)   [18..25] layer2 (same order)
//  [26] mask0 [37,293]  [27] mask1 [24,61]  [28] mask2 [3,27]
// d_out: controls [256*512*3] ++ hx_new [256*64]  (f32)
// ws: [0, 81920) packed f16 B; [81920, +512*256*160*4) f32 Xp (~84 MB)
// ---------------------------------------------------------------------------
extern "C" void kernel_launch(void* const* d_in, const int* in_sizes, int n_in,
                              void* d_out, int out_size, void* d_ws, size_t ws_size,
                              hipStream_t stream) {
    (void)in_sizes; (void)n_in; (void)out_size; (void)ws_size;

    const float* x  = (const float*)d_in[0];
    const float* hx = (const float*)d_in[1];
    const float* P[24];
    for (int i = 0; i < 24; ++i) P[i] = (const float*)d_in[2 + i];
    const float* mask0 = (const float*)d_in[26];
    const float* mask1 = (const float*)d_in[27];
    const float* mask2 = (const float*)d_in[28];

    _Float16* Bp = (_Float16*)d_ws;
    float*    Xp = (float*)((char*)d_ws + BPACK_BYTES);

    float* out_controls = (float*)d_out;
    float* out_hx       = out_controls + (size_t)BATCH * TSTEPS * 3;

    // 1) pack masked layer-0 input weights into WMMA B tiles (f16)
    ncp_pack_b<<<(BPACK_HALVES + 255) / 256, 256, 0, stream>>>(
        P[0], P[2], P[4], P[6], mask0, Bp);

    // 2) hoisted input GEMM: 8192 M-tiles, 1 wave each, 4 waves/block
    ncp_in_gemm<<<2048, 128, 0, stream>>>(x, Bp, Xp);

    // 3) recurrent scan: 64 blocks x 4 batch rows
    ScanArgs A;
    for (int i = 0; i < 8; ++i) { A.w0[i] = P[i]; A.w1[i] = P[8 + i]; A.w2[i] = P[16 + i]; }
    A.mask1 = mask1; A.mask2 = mask2;
    A.hx = hx; A.xp = Xp;
    A.out_controls = out_controls; A.out_hx = out_hx;
    ncp_scan<<<64, STHREADS, 0, stream>>>(A);
}
